// SharedConsciousAgent_75625784148148
// MI455X (gfx1250) — matrix operand order repaired and checked by reading, not compiled, not run on hardware
//
#include <hip/hip_runtime.h>
#include <stdint.h>

// ---------------------------------------------------------------------------
// Types for CDNA5 WMMA (wave32)
// ---------------------------------------------------------------------------
typedef __attribute__((ext_vector_type(16))) __bf16 v16bf;
typedef __attribute__((ext_vector_type(8)))  float  v8f;

struct alignas(16) U4 { uint32_t x, y, z, w; };
union Frag16 { U4 q[2]; uint32_t u[8]; v16bf v; };

__device__ __forceinline__ uint16_t f2bf(float f) {
  uint32_t u = __float_as_uint(f);
  uint32_t r = (u + 0x7FFFu + ((u >> 16) & 1u)) >> 16;
  return (uint16_t)r;
}
__device__ __forceinline__ float bf2f(uint16_t h) {
  return __uint_as_float(((uint32_t)h) << 16);
}

__device__ __forceinline__ float blockReduceSum256(float v, float* sh) {
  int t = threadIdx.x;
  sh[t] = v; __syncthreads();
  for (int s = 128; s > 0; s >>= 1) {
    if (t < s) sh[t] += sh[t + s];
    __syncthreads();
  }
  float r = sh[0]; __syncthreads();
  return r;
}

// dims
#define MTOK 2048   // B*N tokens
#define DD   256
#define DD2  512
#define HH   8192

// ---------------------------------------------------------------------------
// Weight packing: (K x N) row-major f32 -> WMMA B-fragment-major bf16.
// Fragment f = kTile*(N/16)+nTile holds 32 lanes x 16 bf16 contiguous.
// lane<16: n = nTile*16+lane, k = kTile*32 + e  (e=0..15)
// lane>=16: same n, k = kTile*32 + 16 + e
// ---------------------------------------------------------------------------
__global__ void k_packW(const float* __restrict__ src, uint16_t* __restrict__ dst,
                        int K, int N) {
  int idx = blockIdx.x * 256 + threadIdx.x;
  if (idx >= K * N) return;
  int e    = idx & 15;
  int lane = (idx >> 4) & 31;
  int f    = idx >> 9;
  int nT   = f % (N >> 4);
  int kT   = f / (N >> 4);
  int n = nT * 16 + (lane & 15);
  int k = kT * 32 + ((lane >> 4) << 4) + e;
  dst[idx] = f2bf(src[(size_t)k * N + n]);
}

// ---------------------------------------------------------------------------
// Prep: personality modulation + concat(mod, actions) + LN(512) -> bf16.
// Also converts prev_states -> bf16 into gatecat[:, 0:256].
// ---------------------------------------------------------------------------
__global__ void k_prep(const float* __restrict__ sig, const float* __restrict__ actin,
                       const float* __restrict__ prev,
                       const float* __restrict__ pers, const float* __restrict__ chr,
                       const float* __restrict__ g, const float* __restrict__ b,
                       uint16_t* __restrict__ xln, uint16_t* __restrict__ gatecat) {
  __shared__ float sh[256];
  int r = blockIdx.x, t = threadIdx.x, a = r & 63;
  float fp = pers[a * DD + t] + 0.3f * chr[a * DD + t];
  float x0 = sig[(size_t)r * DD + t] * (1.0f + 0.1f * fp);
  float x1 = actin[(size_t)r * DD + t];
  float mu = blockReduceSum256(x0 + x1, sh) * (1.0f / 512.0f);
  float d0 = x0 - mu, d1 = x1 - mu;
  float var = blockReduceSum256(d0 * d0 + d1 * d1, sh) * (1.0f / 512.0f);
  float rs = rsqrtf(var + 1e-5f);
  xln[(size_t)r * DD2 + t]       = f2bf(d0 * rs * g[t] + b[t]);
  xln[(size_t)r * DD2 + DD + t]  = f2bf(d1 * rs * g[DD + t] + b[DD + t]);
  gatecat[(size_t)r * DD2 + t]   = f2bf(prev[(size_t)r * DD + t]);
}

// Decision-input LN: concat(perception f32, recall f32) -> LN(512) -> bf16
__global__ void k_decln(const float* __restrict__ perc, const float* __restrict__ rec,
                        const float* __restrict__ g, const float* __restrict__ b,
                        uint16_t* __restrict__ xln) {
  __shared__ float sh[256];
  int r = blockIdx.x, t = threadIdx.x;
  float x0 = perc[(size_t)r * DD + t];
  float x1 = rec[(size_t)r * DD + t];
  float mu = blockReduceSum256(x0 + x1, sh) * (1.0f / 512.0f);
  float d0 = x0 - mu, d1 = x1 - mu;
  float var = blockReduceSum256(d0 * d0 + d1 * d1, sh) * (1.0f / 512.0f);
  float rs = rsqrtf(var + 1e-5f);
  xln[(size_t)r * DD2 + t]      = f2bf(d0 * rs * g[t] + b[t]);
  xln[(size_t)r * DD2 + DD + t] = f2bf(d1 * rs * g[DD + t] + b[DD + t]);
}

// sim/strength + epcr bf16 emission
__global__ void k_sim(const float* __restrict__ ep, const float* __restrict__ cr,
                      const uint16_t* __restrict__ hdcq, const float* __restrict__ akey,
                      uint16_t* __restrict__ epcr, float* __restrict__ strength) {
  __shared__ float sh[256];
  int r = blockIdx.x, t = threadIdx.x, a = r & 63;
  float part = 0.0f;
  for (int h = t; h < HH; h += 256) {
    size_t i = (size_t)r * HH + h;
    float s = ep[i] + cr[i];
    epcr[i] = f2bf(s);
    part += s * bf2f(hdcq[i]) * akey[(size_t)a * HH + h];
  }
  float sim = blockReduceSum256(part, sh) * (1.0f / (float)HH);
  if (t == 0) strength[r] = 1.0f / (1.0f + expf(-sim));
}

// gated state update + output LN -> new_states (f32 to d_out + bf16 copy)
__global__ void k_state(const float* __restrict__ gate, const float* __restrict__ dec,
                        const float* __restrict__ prev,
                        const float* __restrict__ g, const float* __restrict__ b,
                        float* __restrict__ nsOut, uint16_t* __restrict__ nsB) {
  __shared__ float sh[256];
  int r = blockIdx.x, t = threadIdx.x;
  size_t i = (size_t)r * DD + t;
  float gg = gate[i];
  float s = gg * dec[i] + (1.0f - gg) * prev[i];
  float mu = blockReduceSum256(s, sh) * (1.0f / 256.0f);
  float d = s - mu;
  float var = blockReduceSum256(d * d, sh) * (1.0f / 256.0f);
  float v = d * rsqrtf(var + 1e-5f) * g[t] + b[t];
  nsOut[i] = v;
  nsB[i] = f2bf(v);
}

// mem_gate = sigmoid([new_states, hdc_query] . W_mem + b_mem)
__global__ void k_memgate(const float* __restrict__ ns, const uint16_t* __restrict__ hdcq,
                          const float* __restrict__ wmem, const float* __restrict__ bmem,
                          float* __restrict__ out) {
  __shared__ float sh[256];
  int r = blockIdx.x, t = threadIdx.x;
  float part = ns[(size_t)r * DD + t] * wmem[t];
  for (int h = t; h < HH; h += 256)
    part += bf2f(hdcq[(size_t)r * HH + h]) * wmem[DD + h];
  float s = blockReduceSum256(part, sh);
  if (t == 0) out[r] = 1.0f / (1.0f + expf(-(s + bmem[0])));
}

// ---------------------------------------------------------------------------
// Generic bf16 WMMA GEMM: C = act(A(MxK) * B(KxN) + bias) with fused epilogue.
// One wave computes a 16(M) x 64(N) tile: 4 accumulators, A-frag reuse x4.
// act: 0=none, 1=exact GELU, 2=tanh, 3=sigmoid
// ---------------------------------------------------------------------------
__global__ void __launch_bounds__(256)
k_gemm(const uint16_t* __restrict__ A, const uint16_t* __restrict__ Bp,
       const float* __restrict__ bias,
       float* __restrict__ outF, uint16_t* __restrict__ outB, int ldOutB, int colOffB,
       const float* __restrict__ rowScale, const float* __restrict__ pers,
       const float* __restrict__ ep, const float* __restrict__ akey,
       const float* __restrict__ mg, float* __restrict__ outEpis,
       int M, int K, int N, int act) {
  int wave = (int)((blockIdx.x * blockDim.x + threadIdx.x) >> 5);
  int lane = threadIdx.x & 31;
  int tN4 = N >> 6;
  int tM  = M >> 4;
  if (wave >= tM * tN4) return;
  int tm = wave / tN4, tn = wave % tN4;
  int row0 = tm << 4, n0 = tn << 6;

  v8f acc[4];
#pragma unroll
  for (int t = 0; t < 4; ++t) acc[t] = (v8f){0.f,0.f,0.f,0.f,0.f,0.f,0.f,0.f};

  int aRow = row0 + (lane & 15);
  int aOff = (lane >> 4) << 4;                       // 0 or 16
  const U4* Ar = (const U4*)(A + (size_t)aRow * K);  // 16B units
  int nTiles = N >> 4;
  int nT16base = tn << 2;
  int kTiles = K >> 5;

  for (int kt = 0; kt < kTiles; ++kt) {
    Frag16 af;
    int kb = kt << 5;
    af.q[0] = Ar[(kb + aOff) >> 3];
    af.q[1] = Ar[(kb + 8 + aOff) >> 3];
    const U4* Bq = (const U4*)Bp + (((size_t)kt * nTiles + nT16base) << 6) + (lane << 1);
#pragma unroll
    for (int t = 0; t < 4; ++t) {
      Frag16 bf;
      bf.q[0] = Bq[0];
      bf.q[1] = Bq[1];
      Bq += 64;  // next 16-col fragment (1 KB)
      acc[t] = __builtin_amdgcn_wmma_f32_16x16x32_bf16(
          false, af.v, false, bf.v, (short)0, acc[t], false, false);
    }
  }

  int rbase = row0 + ((lane >> 4) << 3);
  int cb = n0 + (lane & 15);
#pragma unroll
  for (int t = 0; t < 4; ++t) {
    int col = cb + (t << 4);
    float bv = bias ? bias[col] : 0.0f;
#pragma unroll
    for (int v = 0; v < 8; ++v) {
      int row = rbase + v;
      float c = acc[t][v] + bv;
      if (act == 1)      c = 0.5f * c * (1.0f + erff(c * 0.70710678118654752f));
      else if (act == 2) c = tanhf(c);
      else if (act == 3) c = 1.0f / (1.0f + expf(-c));
      if (rowScale) c *= rowScale[row];
      if (pers)     c *= (1.0f + 0.1f * pers[((row & 63) << 8) + col]);  // N==256 path
      size_t oi = (size_t)row * N + col;
      if (outF) outF[oi] = c;
      if (outB) outB[(size_t)row * ldOutB + colOffB + col] = f2bf(c);
      if (outEpis)
        outEpis[oi] = 0.95f * ep[oi] + mg[row] * c * akey[(size_t)(row & 63) * N + col];
    }
  }
}

// ---------------------------------------------------------------------------
extern "C" void kernel_launch(void* const* d_in, const int* in_sizes, int n_in,
                              void* d_out, int out_size, void* d_ws, size_t ws_size,
                              hipStream_t stream) {
  const float* sig    = (const float*)d_in[0];
  const float* actin  = (const float*)d_in[1];
  const float* epi    = (const float*)d_in[2];
  const float* cry    = (const float*)d_in[3];
  const float* prev   = (const float*)d_in[4];
  // d_in[5] = step (unused by the math)
  const float* lnp_g  = (const float*)d_in[6];
  const float* lnp_b  = (const float*)d_in[7];
  const float* W_perc = (const float*)d_in[8];
  const float* b_perc = (const float*)d_in[9];
  const float* lnd_g  = (const float*)d_in[10];
  const float* lnd_b  = (const float*)d_in[11];
  const float* W_dec1 = (const float*)d_in[12];
  const float* b_dec1 = (const float*)d_in[13];
  const float* W_dec2 = (const float*)d_in[14];
  const float* b_dec2 = (const float*)d_in[15];
  const float* W_gate = (const float*)d_in[16];
  const float* b_gate = (const float*)d_in[17];
  const float* W_act  = (const float*)d_in[18];
  const float* b_act  = (const float*)d_in[19];
  const float* W_thdc = (const float*)d_in[20];
  const float* b_thdc = (const float*)d_in[21];
  const float* W_fhdc = (const float*)d_in[22];
  const float* b_fhdc = (const float*)d_in[23];
  const float* W_mem  = (const float*)d_in[24];
  const float* b_mem  = (const float*)d_in[25];
  const float* pers   = (const float*)d_in[26];
  const float* chr    = (const float*)d_in[27];
  const float* akey   = (const float*)d_in[28];
  const float* lno_g  = (const float*)d_in[29];
  const float* lno_b  = (const float*)d_in[30];

  // Output sections (float32, flat in return order)
  float* out       = (float*)d_out;
  float* o_actions = out;
  float* o_ns      = o_actions + (size_t)MTOK * DD;
  float* o_epis    = o_ns      + (size_t)MTOK * DD;
  float* o_exp     = o_epis    + (size_t)MTOK * HH;
  float* o_mg      = o_exp     + (size_t)MTOK * HH;

  // Workspace layout
  char* ws = (char*)d_ws;
  size_t off = 0;
  auto alloc = [&](size_t bytes) -> char* {
    char* p = ws + off;
    off += (bytes + 255) & ~(size_t)255;
    return p;
  };
  uint16_t* wPercP = (uint16_t*)alloc((size_t)DD2 * DD * 2);
  uint16_t* wDec1P = (uint16_t*)alloc((size_t)DD2 * DD * 2);
  uint16_t* wDec2P = (uint16_t*)alloc((size_t)DD  * DD * 2);
  uint16_t* wGateP = (uint16_t*)alloc((size_t)DD2 * DD * 2);
  uint16_t* wActP  = (uint16_t*)alloc((size_t)DD  * DD * 2);
  uint16_t* wThdcP = (uint16_t*)alloc((size_t)DD  * HH * 2);
  uint16_t* wFhdcP = (uint16_t*)alloc((size_t)HH  * DD * 2);
  uint16_t* xlnB   = (uint16_t*)alloc((size_t)MTOK * DD2 * 2);
  float*    percF  = (float*)   alloc((size_t)MTOK * DD * 4);
  uint16_t* percB  = (uint16_t*)alloc((size_t)MTOK * DD * 2);
  uint16_t* hdcqB  = (uint16_t*)alloc((size_t)MTOK * HH * 2);
  uint16_t* epcrB  = (uint16_t*)alloc((size_t)MTOK * HH * 2);
  float*    strg   = (float*)   alloc((size_t)MTOK * 4);
  float*    recF   = (float*)   alloc((size_t)MTOK * DD * 4);
  uint16_t* declnB = (uint16_t*)alloc((size_t)MTOK * DD2 * 2);
  uint16_t* hB     = (uint16_t*)alloc((size_t)MTOK * DD * 2);
  float*    decF   = (float*)   alloc((size_t)MTOK * DD * 4);
  uint16_t* gcatB  = (uint16_t*)alloc((size_t)MTOK * DD2 * 2);
  float*    gateF  = (float*)   alloc((size_t)MTOK * DD * 4);
  uint16_t* nsB    = (uint16_t*)alloc((size_t)MTOK * DD * 2);
  (void)ws_size; (void)in_sizes; (void)n_in; (void)out_size;

  auto pack = [&](const float* s, uint16_t* dct, int K, int N) {
    int n = K * N;
    k_packW<<<(n + 255) / 256, 256, 0, stream>>>(s, dct, K, N);
  };
  auto gemm = [&](const uint16_t* Aa, const uint16_t* Bb, const float* bb,
                  float* oF, uint16_t* oB, int ldB, int coff,
                  const float* rs, const float* ps,
                  const float* epp, const float* ak, const float* mgp, float* oE,
                  int M, int K, int N, int act) {
    int tiles = (M >> 4) * (N >> 6);
    int blocks = (tiles + 7) >> 3;  // 8 waves / block
    k_gemm<<<blocks, 256, 0, stream>>>(Aa, Bb, bb, oF, oB, ldB, coff,
                                       rs, ps, epp, ak, mgp, oE, M, K, N, act);
  };

  // 1) pack weights to fragment-major bf16
  pack(W_perc, wPercP, DD2, DD);
  pack(W_dec1, wDec1P, DD2, DD);
  pack(W_dec2, wDec2P, DD,  DD);
  pack(W_gate, wGateP, DD2, DD);
  pack(W_act,  wActP,  DD,  DD);
  pack(W_thdc, wThdcP, DD,  HH);
  pack(W_fhdc, wFhdcP, HH,  DD);

  // 2) modulation + perception LN + prev_states->bf16
  k_prep<<<MTOK, 256, 0, stream>>>(sig, actin, prev, pers, chr, lnp_g, lnp_b, xlnB, gcatB);

  // 3) perception = GELU(xln @ W_perc + b)   (f32 + bf16 copies)
  gemm(xlnB, wPercP, b_perc, percF, percB, DD, 0,
       nullptr, nullptr, nullptr, nullptr, nullptr, nullptr, MTOK, DD2, DD, 1);

  // 4) hdc_query = tanh(perception @ W_to_hdc + b)   (bf16)
  gemm(percB, wThdcP, b_thdc, nullptr, hdcqB, HH, 0,
       nullptr, nullptr, nullptr, nullptr, nullptr, nullptr, MTOK, DD, HH, 2);

  // 5) sim/strength; emit epcr = (ep+cr) in bf16
  k_sim<<<MTOK, 256, 0, stream>>>(epi, cry, hdcqB, akey, epcrB, strg);

  // 6) recall = ((ep+cr) @ W_from_hdc + b) * strength
  gemm(epcrB, wFhdcP, b_fhdc, recF, nullptr, 0, 0,
       strg, nullptr, nullptr, nullptr, nullptr, nullptr, MTOK, HH, DD, 0);

  // 7) decision-input LN
  k_decln<<<MTOK, 256, 0, stream>>>(percF, recF, lnd_g, lnd_b, declnB);

  // 8) h = GELU(decln @ W_dec1 + b)
  gemm(declnB, wDec1P, b_dec1, nullptr, hB, DD, 0,
       nullptr, nullptr, nullptr, nullptr, nullptr, nullptr, MTOK, DD2, DD, 1);

  // 9) decision = (h @ W_dec2 + b) * (1 + 0.1*personality); also fill gatecat[:,256:]
  gemm(hB, wDec2P, b_dec2, decF, gcatB, DD2, DD,
       nullptr, pers, nullptr, nullptr, nullptr, nullptr, MTOK, DD, DD, 0);

  // 10) gate = sigmoid([prev_states, decision] @ W_gate + b)
  gemm(gcatB, wGateP, b_gate, gateF, nullptr, 0, 0,
       nullptr, nullptr, nullptr, nullptr, nullptr, nullptr, MTOK, DD2, DD, 3);

  // 11) new_states = LN(gate*decision + (1-gate)*prev)   (-> d_out + bf16)
  k_state<<<MTOK, 256, 0, stream>>>(gateF, decF, prev, lno_g, lno_b, o_ns, nsB);

  // 12) actions = new_states @ W_act + b   (-> d_out)
  gemm(nsB, wActP, b_act, o_actions, nullptr, 0, 0,
       nullptr, nullptr, nullptr, nullptr, nullptr, nullptr, MTOK, DD, DD, 0);

  // 13) mem_gate = sigmoid([new_states, hdc_query] . W_mem + b)   (-> d_out)
  k_memgate<<<MTOK, 256, 0, stream>>>(o_ns, hdcqB, W_mem, b_mem, o_mg);

  // 14) exp_hdc = tanh(new_states @ W_to_hdc + b)  -> d_out
  //     fused: new_episodics = 0.95*ep + mem_gate * exp_hdc * agent_key -> d_out
  gemm(nsB, wThdcP, b_thdc, o_exp, nullptr, 0, 0,
       nullptr, nullptr, epi, akey, o_mg, o_epis, MTOK, DD, HH, 2);
}
